// GRU_6691559047735
// MI455X (gfx1250) — matrix operand order repaired and checked by reading, not compiled, not run on hardware
//
#include <hip/hip_runtime.h>
#include <hip/hip_fp16.h>

// GRU (2 layers, T=2000, B=512, INP=40, H=20) for MI455X / gfx1250.
//
//  K1: proj_kernel<40> : wz/wr/wh = x @ W^T + b for all T*B rows  (WMMA f16->f32)
//  K2: gru_scan        : layer-0 recurrence, 32 independent waves (1 per 16 batch
//                        rows), 2000 steps, 6 WMMAs/step, h kept in regs+LDS,
//                        projections streamed through a 5-deep LDS ring filled
//                        with global_load_async_to_lds_b128 (ASYNCcnt-paced)
//  K3: proj_kernel<20> : layer-1 projections from h0 (f16)
//  K4: gru_scan        : layer-1 recurrence -> d_out (f32)

#define T_STEPS 2000
#define BATCH   512
#define INPD    40
#define HD      20
#define MT      (BATCH / 16)          // 32 M-tiles of the batch
#define NTI     2                     // N tiles per gate (20 cols -> 32 padded)
#define TILE_F32 256                  // f32 per stored D tile (32 lanes * 8)
#define STEP_F32 (3 * NTI * TILE_F32) // 1536 f32 per (t, mtile): z|r|h tiles
#define STEP_BYTES (STEP_F32 * 4)     // 6144 B
#define RING_D  5                     // ring depth: 5*12=60 async ops in flight <= 63

typedef __attribute__((ext_vector_type(16))) _Float16 v16h;
typedef __attribute__((ext_vector_type(8)))  float    v8f;

// ---- WMMA fragment index maps (CDNA5 ISA, 16-bit layouts) ----
__device__ __forceinline__ int a_kmap(int e, int hi) {
  // A 16x32 f16: lanes0-15: K = e(0..7), e+8(8..15); lanes16-31: +8
  return (e < 8 ? e : e + 8) + (hi ? 8 : 0);
}
__device__ __forceinline__ int b_kmap(int e, int hi) {
  // B 32x16 f16: lanes0-15: K = e(0..7), e+8(8..15); lanes16-31: +8
  return (e < 8 ? e : e + 8) + (hi ? 8 : 0);
}

__device__ __forceinline__ v8f wmma16(v16h a, v16h b, v8f c) {
  return __builtin_amdgcn_wmma_f32_16x16x32_f16(false, a, false, b,
                                                (short)0, c, false, false);
}

__device__ __forceinline__ v8f load_c8(const float* p) {
  const float4* q = (const float4*)p;
  float4 a = q[0], b = q[1];
  v8f c;
  c[0]=a.x; c[1]=a.y; c[2]=a.z; c[3]=a.w;
  c[4]=b.x; c[5]=b.y; c[6]=b.z; c[7]=b.w;
  return c;
}
__device__ __forceinline__ void store_c8(float* p, v8f c) {
  float4 a, b;
  a.x=c[0]; a.y=c[1]; a.z=c[2]; a.w=c[3];
  b.x=c[4]; b.y=c[5]; b.z=c[6]; b.w=c[7];
  ((float4*)p)[0] = a; ((float4*)p)[1] = b;
}

__device__ __forceinline__ float fast_sigmoid(float x) {
  return __builtin_amdgcn_rcpf(1.0f + __expf(-x));
}
__device__ __forceinline__ float fast_tanh(float x) {
  float e2 = __expf(-2.0f * x);
  return (1.0f - e2) * __builtin_amdgcn_rcpf(1.0f + e2);
}

// Async-fill one 6 KB step block (12 x b128 per lane) into the LDS ring.
// GVS addressing: saddr = uniform block base, vaddr = per-lane byte offset.
__device__ __forceinline__ void async_fill(const float* gsrc, unsigned ldsoff,
                                           int lane) {
  unsigned long long base = (unsigned long long)gsrc;
  #pragma unroll
  for (int i = 0; i < 12; ++i) {
    unsigned voff = (unsigned)((i * 32 + lane) * 16);
    unsigned loff = ldsoff + voff;
    asm volatile("global_load_async_to_lds_b128 %0, %1, %2"
                 :: "v"(loff), "v"(voff), "s"(base) : "memory");
  }
}
// Wait until at most 4 step blocks (48 async ops) remain outstanding.
__device__ __forceinline__ void async_wait_4blocks() {
  asm volatile("s_wait_asynccnt 48" ::: "memory");
}

// ---------------------------------------------------------------------------
// Projection GEMM: out[t,b,gate,:] = in[t,b,:] @ W_gate^T + b_gate
// One wave per 16-row M-tile. KIN = 40 (f32 x) or 20 (f16 h0).
// ---------------------------------------------------------------------------
template <int KIN, bool A_IS_F16>
__global__ __launch_bounds__(256) void proj_kernel(
    const void* __restrict__ ain,
    const float* __restrict__ Wz, const float* __restrict__ Wr,
    const float* __restrict__ Wh,
    const float* __restrict__ bz, const float* __restrict__ br,
    const float* __restrict__ bh,
    float* __restrict__ proj)
{
  constexpr int KC = (KIN + 31) / 32;           // K chunks of 32
  const int lane = threadIdx.x & 31;
  const int hi   = lane >> 4;
  const int ln   = lane & 15;
  const int wave = blockIdx.x * (blockDim.x >> 5) + (threadIdx.x >> 5);
  const int n_mtiles = (T_STEPS * BATCH) / 16;  // 64000
  if (wave >= n_mtiles) return;

  const float* W[3]  = {Wz, Wr, Wh};
  const float* Bb[3] = {bz, br, bh};

  // Build B fragments (weights, packed to f16) once per wave.
  v16h bf[3][NTI][KC];
  #pragma unroll
  for (int g = 0; g < 3; ++g)
    #pragma unroll
    for (int nt = 0; nt < NTI; ++nt)
      #pragma unroll
      for (int kc = 0; kc < KC; ++kc) {
        v16h f;
        #pragma unroll
        for (int e = 0; e < 16; ++e) {
          int k = b_kmap(e, hi) + 32 * kc;
          int n = nt * 16 + ln;
          float w = (k < KIN && n < HD) ? W[g][n * KIN + (k < KIN ? k : 0)] : 0.0f;
          f[e] = (_Float16)w;
        }
        bf[g][nt][kc] = f;
      }

  // Bias (per column, broadcast over the 8 row-slots of a C fragment).
  float bias[3][NTI];
  #pragma unroll
  for (int g = 0; g < 3; ++g)
    #pragma unroll
    for (int nt = 0; nt < NTI; ++nt) {
      int n = nt * 16 + ln;
      bias[g][nt] = (n < HD) ? Bb[g][n] : 0.0f;
    }

  // A fragments for this 16-row tile.
  const long row = (long)wave * 16 + ln;
  v16h af[KC];
  #pragma unroll
  for (int kc = 0; kc < KC; ++kc) {
    v16h f;
    #pragma unroll
    for (int e = 0; e < 16; ++e) {
      int k = a_kmap(e, hi) + 32 * kc;
      int ks = (k < KIN) ? k : 0;
      float v;
      if constexpr (A_IS_F16) v = (float)((const _Float16*)ain)[row * KIN + ks];
      else                    v = ((const float*)ain)[row * KIN + ks];
      f[e] = (k < KIN) ? (_Float16)v : (_Float16)0.0f;
    }
    af[kc] = f;
  }

  // D = A*B + bias, stored in fragment order for the scan.
  float* obase = proj + (long)wave * STEP_F32;
  #pragma unroll
  for (int g = 0; g < 3; ++g)
    #pragma unroll
    for (int nt = 0; nt < NTI; ++nt) {
      v8f acc;
      #pragma unroll
      for (int v = 0; v < 8; ++v) acc[v] = bias[g][nt];
      #pragma unroll
      for (int kc = 0; kc < KC; ++kc)
        acc = wmma16(af[kc], bf[g][nt][kc], acc);
      store_c8(obase + ((g * NTI + nt) * TILE_F32) + lane * 8, acc);
    }
}

// ---------------------------------------------------------------------------
// Recurrent scan: one wave per 16 batch rows, loops all T steps.
//   z = sig(wz + h Uz^T); r = sig(wr + h Ur^T)
//   hc = tanh(wh + (r*h) Uh^T); h = z*h + (1-z)*hc
// Projections streamed through an LDS ring via async loads.
// ---------------------------------------------------------------------------
__global__ __launch_bounds__(32) void gru_scan(
    const float* __restrict__ proj,
    const float* __restrict__ Uz, const float* __restrict__ Ur,
    const float* __restrict__ Uh,
    void* __restrict__ outp, int out_f32)
{
  const int lane = threadIdx.x & 31;
  const int hi   = lane >> 4;
  const int ln   = lane & 15;
  const int mt   = blockIdx.x;                  // 0..31

  __shared__ float    ring[RING_D * STEP_F32];  // 30 KB async staging ring
  __shared__ _Float16 hstage[16 * HD];          // canonical h (f16) for A reads
  __shared__ _Float16 rhstage[16 * HD];         // r*h staging

  const unsigned ring0 = (unsigned)(unsigned long long)(void*)&ring[0];

  // Recurrent weight B fragments (built once, live in regs for all 2000 steps).
  const float* U[3] = {Uz, Ur, Uh};
  v16h uf[3][NTI];
  #pragma unroll
  for (int g = 0; g < 3; ++g)
    #pragma unroll
    for (int nt = 0; nt < NTI; ++nt) {
      v16h f;
      #pragma unroll
      for (int e = 0; e < 16; ++e) {
        int k = b_kmap(e, hi);
        int n = nt * 16 + ln;
        float w = (k < HD && n < HD) ? U[g][n * HD + (k < HD ? k : 0)] : 0.0f;
        f[e] = (_Float16)w;
      }
      uf[g][nt] = f;
    }

  // h = 0
  v8f hD[NTI];
  #pragma unroll
  for (int nt = 0; nt < NTI; ++nt)
    #pragma unroll
    for (int v = 0; v < 8; ++v) hD[nt][v] = 0.0f;
  for (int i = lane; i < 16 * HD; i += 32) hstage[i] = (_Float16)0.0f;
  __syncthreads();

  // Prime the ring with steps 0..RING_D-1.
  #pragma unroll
  for (int d = 0; d < RING_D; ++d)
    async_fill(proj + (long)(d * MT + mt) * STEP_F32,
               ring0 + (unsigned)(d * STEP_BYTES), lane);

  int slot = 0;
  for (int t = 0; t < T_STEPS; ++t) {
    // Ensure step t's block has landed in LDS.
    async_wait_4blocks();
    const float* cbase = ring + slot * STEP_F32;

    // A fragment of h from LDS stage.
    v16h ah;
    #pragma unroll
    for (int e = 0; e < 16; ++e) {
      int k = a_kmap(e, hi);
      _Float16 v = hstage[ln * HD + (k < HD ? k : 0)];
      ah[e] = (k < HD) ? v : (_Float16)0.0f;
    }

    // z and r gates.
    v8f zD[NTI], rD[NTI];
    #pragma unroll
    for (int nt = 0; nt < NTI; ++nt)
      zD[nt] = wmma16(ah, uf[0][nt], load_c8(cbase + (0 * NTI + nt) * TILE_F32 + lane * 8));
    #pragma unroll
    for (int nt = 0; nt < NTI; ++nt)
      rD[nt] = wmma16(ah, uf[1][nt], load_c8(cbase + (1 * NTI + nt) * TILE_F32 + lane * 8));
    #pragma unroll
    for (int nt = 0; nt < NTI; ++nt)
      #pragma unroll
      for (int v = 0; v < 8; ++v) {
        zD[nt][v] = fast_sigmoid(zD[nt][v]);
        rD[nt][v] = fast_sigmoid(rD[nt][v]);
      }

    // r*h -> LDS (f16) -> A fragment.
    #pragma unroll
    for (int nt = 0; nt < NTI; ++nt)
      #pragma unroll
      for (int v = 0; v < 8; ++v) {
        int col = nt * 16 + ln;
        int rrow = v + 8 * hi;
        if (col < HD) rhstage[rrow * HD + col] = (_Float16)(rD[nt][v] * hD[nt][v]);
      }
    __syncthreads();
    v16h arh;
    #pragma unroll
    for (int e = 0; e < 16; ++e) {
      int k = a_kmap(e, hi);
      _Float16 v = rhstage[ln * HD + (k < HD ? k : 0)];
      arh[e] = (k < HD) ? v : (_Float16)0.0f;
    }

    // Candidate state + h update.
    v8f hc[NTI];
    #pragma unroll
    for (int nt = 0; nt < NTI; ++nt)
      hc[nt] = wmma16(arh, uf[2][nt], load_c8(cbase + (2 * NTI + nt) * TILE_F32 + lane * 8));
    #pragma unroll
    for (int nt = 0; nt < NTI; ++nt)
      #pragma unroll
      for (int v = 0; v < 8; ++v) {
        float th = fast_tanh(hc[nt][v]);
        hD[nt][v] = zD[nt][v] * hD[nt][v] + (1.0f - zD[nt][v]) * th;
      }

    // Refill the slot we just consumed with step t+RING_D (after the ds_loads
    // above have been waited on by their WMMA consumers).
    if (t + RING_D < T_STEPS)
      async_fill(proj + (long)((t + RING_D) * MT + mt) * STEP_F32,
                 ring0 + (unsigned)(slot * STEP_BYTES), lane);

    // Publish h (LDS f16 for next step's A read) and emit h_t.
    __syncthreads();
    #pragma unroll
    for (int nt = 0; nt < NTI; ++nt)
      #pragma unroll
      for (int v = 0; v < 8; ++v) {
        int col = nt * 16 + ln;
        int rrow = v + 8 * hi;
        if (col < HD) {
          float hv = hD[nt][v];
          hstage[rrow * HD + col] = (_Float16)hv;
          long gidx = ((long)t * BATCH + mt * 16 + rrow) * HD + col;
          if (out_f32) ((float*)outp)[gidx] = hv;
          else         ((_Float16*)outp)[gidx] = (_Float16)hv;
        }
      }
    __syncthreads();

    slot = (slot + 1 == RING_D) ? 0 : slot + 1;
  }
}

// ---------------------------------------------------------------------------
extern "C" void kernel_launch(void* const* d_in, const int* in_sizes, int n_in,
                              void* d_out, int out_size, void* d_ws, size_t ws_size,
                              hipStream_t stream) {
  (void)in_sizes; (void)n_in; (void)out_size; (void)ws_size;
  const float* x   = (const float*)d_in[0];
  const float* Wh0 = (const float*)d_in[1];
  const float* bh0 = (const float*)d_in[2];
  const float* Wz0 = (const float*)d_in[3];
  const float* bz0 = (const float*)d_in[4];
  const float* Wr0 = (const float*)d_in[5];
  const float* br0 = (const float*)d_in[6];
  const float* Uh0 = (const float*)d_in[7];
  const float* Uz0 = (const float*)d_in[8];
  const float* Ur0 = (const float*)d_in[9];
  const float* Wh1 = (const float*)d_in[10];
  const float* bh1 = (const float*)d_in[11];
  const float* Wz1 = (const float*)d_in[12];
  const float* bz1 = (const float*)d_in[13];
  const float* Wr1 = (const float*)d_in[14];
  const float* br1 = (const float*)d_in[15];
  const float* Uh1 = (const float*)d_in[16];
  const float* Uz1 = (const float*)d_in[17];
  const float* Ur1 = (const float*)d_in[18];

  // Workspace: projection fragments (f32) then h0 (f16).
  float* proj = (float*)d_ws;
  const size_t projFloats = (size_t)T_STEPS * MT * STEP_F32;   // ~98.3M f32
  _Float16* h0 = (_Float16*)((char*)d_ws + projFloats * sizeof(float));

  const int n_mtiles = (T_STEPS * BATCH) / 16;                 // 64000
  dim3 blk(256);
  dim3 grd(n_mtiles / 8);                                      // 8 waves/block

  // Layer 0
  proj_kernel<INPD, false><<<grd, blk, 0, stream>>>(x, Wz0, Wr0, Wh0,
                                                    bz0, br0, bh0, proj);
  gru_scan<<<MT, 32, 0, stream>>>(proj, Uz0, Ur0, Uh0, (void*)h0, 0);
  // Layer 1
  proj_kernel<HD, true><<<grd, blk, 0, stream>>>(h0, Wz1, Wr1, Wh1,
                                                 bz1, br1, bh1, proj);
  gru_scan<<<MT, 32, 0, stream>>>(proj, Uz1, Ur1, Uh1, d_out, 1);
}
